// LSTM_75402445849218
// MI455X (gfx1250) — matrix operand order repaired and checked by reading, not compiled, not run on hardware
//
#include <hip/hip_runtime.h>

#define T_SEQ   512
#define BATCHSZ 4096
#define HID     100
#define G4      400      // 4*HID gate columns
#define BM      16       // batch rows per workgroup
#define NWAVES  5        // 5 waves x 5 N-tiles = 25 tiles, perfectly uniform
#define TPT     5        // tiles per wave
#define BLOCK   (NWAVES * 32)
#define HSTRIDE 136      // bf16 elems per h row: 128 padded-K + 8 pad (68 dwords -> conflict-free)
#define GSTRIDE 404      // f32 elems per gates row (404 mod 64 = 20 -> conflict-free column access)

typedef __attribute__((ext_vector_type(16))) __bf16 v16bf;
typedef __attribute__((ext_vector_type(8)))  __bf16 v8bf;
typedef __attribute__((ext_vector_type(8)))  float  v8f;

__device__ __forceinline__ float fast_sigmoid(float x) {
    return 1.0f / (1.0f + __expf(-x));
}
__device__ __forceinline__ float fast_tanh(float x) {
    return 2.0f / (1.0f + __expf(-2.0f * x)) - 1.0f;
}

__global__ __launch_bounds__(BLOCK)
void LSTM_75402445849218_kernel(const float* __restrict__ x,
                                const float* __restrict__ Wih,
                                const float* __restrict__ Whh,
                                const float* __restrict__ bih,
                                const float* __restrict__ bhh,
                                const float* __restrict__ Wlin,
                                const float* __restrict__ blin,
                                float* __restrict__ out) {
    __shared__ __align__(16) __bf16 ldsH[BM * HSTRIDE];  // h_{t-1}, bf16, K-padded w/ zeros
    __shared__ float ldsG[BM * GSTRIDE];                 // gate pre-activations (f32)
    __shared__ float ldsC[BM * HID];                     // cell state, [n][m] layout (e = n*16+m)

    const int tid     = threadIdx.x;
    const int wave    = tid >> 5;       // 0..4
    const int lane    = tid & 31;
    const int lhalf   = lane >> 4;      // lane group 0/1 (wave32 WMMA striping)
    const int lmod    = lane & 15;
    const int rowBase = blockIdx.x * BM;

    // Uniform tiling: wave w owns N-tiles w*5 .. w*5+4 (gate cols w*80 .. w*80+79)
    // ---- preload register-resident B fragments (W_hh as bf16), bias, W_ih ----
    // B layout (16x16x32 bf16): lane = (n = lmod, kgroup = lhalf), halves j -> K = ks*32 + lhalf*16 + j
    v16bf Bfrag[TPT][4];                // [tile][kstep]
    float biasReg[TPT], wihReg[TPT];
    #pragma unroll
    for (int ti = 0; ti < TPT; ++ti) {
        const int n = (wave * TPT + ti) * 16 + lmod;   // gate column 0..399
        biasReg[ti] = bih[n] + bhh[n];
        wihReg[ti]  = Wih[n];
        const float* wrow = Whh + n * HID;             // W_hh[n][k], k contiguous
        #pragma unroll
        for (int ks = 0; ks < 4; ++ks) {
            #pragma unroll
            for (int j = 0; j < 16; ++j) {
                const int k = ks * 32 + lhalf * 16 + j;
                Bfrag[ti][ks][j] = (k < HID) ? (__bf16)wrow[k] : (__bf16)0.0f;
            }
        }
    }

    // ---- init on-chip state: h = 0 (incl. K padding, kept zero forever), c = 0 ----
    for (int i = tid; i < BM * HSTRIDE; i += BLOCK) ldsH[i] = (__bf16)0.0f;
    for (int i = tid; i < BM * HID;     i += BLOCK) ldsC[i] = 0.0f;
    __syncthreads();

    for (int t = 0; t < T_SEQ; ++t) {
        // x_t for this block's 16 batch rows (INPUT_SIZE == 1); L2-resident
        const float* xt = x + (size_t)t * BATCHSZ + rowBase + 8 * lhalf;
        float xv[8];
        #pragma unroll
        for (int r = 0; r < 8; ++r) xv[r] = xt[r];

        v8f acc[TPT];
        #pragma unroll
        for (int ti = 0; ti < TPT; ++ti) { v8f z = {}; acc[ti] = z; }

        // ---- gates = h_{t-1} @ W_hh^T via bf16 WMMA, f32 accumulate ----
        // K-step outer / tile inner: 5 independent WMMAs between dependent accumulates
        const __bf16* ah = ldsH + lmod * HSTRIDE + lhalf * 16;  // single base, imm offsets
        #pragma unroll
        for (int ks = 0; ks < 4; ++ks) {
            // A fragment (16x32 bf16): lane m = lmod; halves j=0..15 -> K = ks*32 + lhalf*16 + j
            union { v16bf v16; v8bf v8[2]; } a;
            a.v8[0] = *(const v8bf*)(ah + ks * 32);
            a.v8[1] = *(const v8bf*)(ah + ks * 32 + 8);
            #pragma unroll
            for (int ti = 0; ti < TPT; ++ti) {
                acc[ti] = __builtin_amdgcn_wmma_f32_16x16x32_bf16(
                    false, a.v16, false, Bfrag[ti][ks],
                    (short)0, acc[ti], false, false);
            }
        }

        // ---- epilogue: + bias + x_t * W_ih, stage gates to LDS ----
        // C/D layout: VGPR r, lanes 0-15 -> M=r, lanes 16-31 -> M=r+8; N = lmod.
        // One base VGPR; r*GSTRIDE and ti*16 are compile-time immediate DS offsets.
        {
            float* gout = ldsG + (lhalf * 8) * GSTRIDE + wave * (TPT * 16) + lmod;
            #pragma unroll
            for (int ti = 0; ti < TPT; ++ti) {
                #pragma unroll
                for (int r = 0; r < 8; ++r) {
                    gout[r * GSTRIDE + ti * 16] = acc[ti][r] + biasReg[ti] + xv[r] * wihReg[ti];
                }
            }
        }
        __syncthreads();

        // ---- elementwise LSTM cell update: 16 rows x 100 hidden units ----
        // e = tid + i*160; 160 % 16 == 0 -> row m = tid&15 is loop-invariant,
        // n = (tid>>4) + 10*i. All accesses are immediate offsets off fixed bases.
        {
            const int m  = tid & 15;    // batch row (fixed per thread)
            const int n0 = tid >> 4;    // starting hidden unit (0..9)
            const float* gr = ldsG + m * GSTRIDE + n0;
            float*       cp = ldsC + n0 * 16 + m;       // e = n*16 + m
            __bf16*      hp = ldsH + m * HSTRIDE + n0;
            #pragma unroll
            for (int i = 0; i < 10; ++i) {              // n = n0 + 10*i < 100
                const float ig = fast_sigmoid(gr[10 * i]);
                const float fg = fast_sigmoid(gr[HID + 10 * i]);
                const float gg = fast_tanh(gr[2 * HID + 10 * i]);
                const float og = fast_sigmoid(gr[3 * HID + 10 * i]);
                const float c  = fg * cp[160 * i] + ig * gg;
                cp[160 * i] = c;
                hp[10 * i] = (__bf16)(og * fast_tanh(c));  // cols >= HID stay zero (K padding)
            }
        }
        __syncthreads();
    }

    // ---- final head: out[b] = h_T[b,:] . W_lin + b_lin (h read back from bf16 state) ----
    if (tid < BM) {
        float s = blin[0];
        const __bf16* hrow = ldsH + tid * HSTRIDE;
        #pragma unroll 4
        for (int n = 0; n < HID; ++n) s += (float)hrow[n] * Wlin[n];
        out[rowBase + tid] = s;
    }
}

extern "C" void kernel_launch(void* const* d_in, const int* in_sizes, int n_in,
                              void* d_out, int out_size, void* d_ws, size_t ws_size,
                              hipStream_t stream) {
    const float* x    = (const float*)d_in[0];   // input_seq [512,4096,1]
    const float* Wih  = (const float*)d_in[1];   // [400,1]
    const float* Whh  = (const float*)d_in[2];   // [400,100]
    const float* bih  = (const float*)d_in[3];   // [400]
    const float* bhh  = (const float*)d_in[4];   // [400]
    const float* Wlin = (const float*)d_in[5];   // [1,100]
    const float* blin = (const float*)d_in[6];   // [1]
    float* out = (float*)d_out;                  // [4096,1]

    dim3 grid(BATCHSZ / BM), block(BLOCK);
    hipLaunchKernelGGL(LSTM_75402445849218_kernel, grid, block, 0, stream,
                       x, Wih, Whh, bih, bhh, Wlin, blin, out);
}
